// DotProductAttention_46634754900131
// MI455X (gfx1250) — matrix operand order repaired and checked by reading, not compiled, not run on hardware
//
#include <hip/hip_runtime.h>
#include <stdint.h>

typedef __attribute__((ext_vector_type(16))) __bf16        v16bf;
typedef __attribute__((ext_vector_type(8)))  float         v8f;
typedef __attribute__((ext_vector_type(8)))  unsigned int  v8u;
typedef __attribute__((ext_vector_type(4)))  unsigned int  v4u;
typedef __attribute__((ext_vector_type(8)))  int           v8i;
typedef __attribute__((ext_vector_type(4)))  int           v4i;

namespace {
constexpr int kB = 16;
constexpr int kQ = 2048;
constexpr int kK = 2048;
constexpr int kD = 64;
constexpr int kQTile   = 64;            // query rows per workgroup
constexpr int kKTile   = 32;            // keys per inner step
constexpr int kWaves   = 4;             // wave32 -> 128 threads
constexpr int kThreads = kWaves * 32;
constexpr int SKS = 72;                 // sK row stride (bf16), padded vs 64
constexpr int SVS = 36;                 // sVt row stride (bf16), padded vs 32
constexpr float kScale  = 0.125f;       // 1/sqrt(64)
constexpr float kLog2e  = 1.44269504088896340736f;
constexpr float kNegInf = -1e30f;       // matches reference NEG_INF
}

// Issue one TDM descriptor: 2D byte tile (tile_dim0 x tile_dim1), strided rows,
// global -> LDS.  Built per CDNA5 ISA (D# groups 0/1).
__device__ __forceinline__ void tdm_load_2d_bytes(unsigned lds_off, uint64_t gaddr,
                                                  unsigned tile_w, unsigned tile_h,
                                                  unsigned row_stride,
                                                  unsigned tdim0, unsigned tdim1) {
  v4u g0;
  g0[0] = 1u;                                        // count=1 (valid), user mode
  g0[1] = lds_off;                                   // lds_addr[63:32]
  g0[2] = (unsigned)(gaddr & 0xFFFFFFFFull);         // global_addr lo
  g0[3] = (unsigned)((gaddr >> 32) & 0x01FFFFFFu)    // global_addr[56:32]
          | (2u << 30);                              // type=2 ("image")
  v8i g1;
  g1[0] = 0;                                         // wg_mask=0, data_size=0 (1B)
  g1[1] = (int)((tdim0 & 0xFFFFu) << 16);            // tensor_dim0[15:0] @ bits63:48
  g1[2] = (int)((tdim0 >> 16) | ((tdim1 & 0xFFFFu) << 16));
  g1[3] = (int)((tdim1 >> 16) | (tile_w << 16));     // tile_dim0 @ bits127:112
  g1[4] = (int)(tile_h);                             // tile_dim1 (tile_dim2=0)
  g1[5] = (int)row_stride;                           // tensor_dim0_stride lo32
  g1[6] = 0;                                         // stride hi16, dim1_stride lo16
  g1[7] = 0;
  v4i z4 = {0, 0, 0, 0};
#if defined(__clang_major__) && (__clang_major__ >= 23)
  v8i z8 = {0, 0, 0, 0, 0, 0, 0, 0};
  __builtin_amdgcn_tensor_load_to_lds(g0, g1, z4, z4, z8, 0);
#else
  __builtin_amdgcn_tensor_load_to_lds(g0, g1, z4, z4, 0);
#endif
}

__global__ __launch_bounds__(kThreads)
void fattn_fwd(const float* __restrict__ q,
               const float* __restrict__ k,
               const float* __restrict__ v,
               const uint8_t* __restrict__ mask,
               float* __restrict__ out) {
  __shared__ __align__(16) __bf16  sK[kKTile * SKS];        // K tile [key][d] bf16
  __shared__ __align__(16) __bf16  sVt[kD * SVS];           // V tile [d][key] bf16
  __shared__ __align__(16) __bf16  sP[kWaves][16 * kKTile]; // per-wave P tile
  __shared__ __align__(16) uint8_t sM[kQTile * kKTile];     // mask tile via TDM

  const int tid  = threadIdx.x;
  const int lane = tid & 31;
  const int wave = tid >> 5;
  const int h    = lane >> 4;   // half-wave select (WMMA layout)
  const int l16  = lane & 15;

  const int b   = blockIdx.x / (kQ / kQTile);
  const int qt  = blockIdx.x % (kQ / kQTile);
  const int qwg = qt * kQTile;              // workgroup's first query row
  const int q0  = qwg + wave * 16;          // this wave's 16 query rows

  // ---- Q tile (16x64) -> bf16 A-matrix layout, two 16x32 k-steps ----
  const float* qrow = q + ((size_t)b * kQ + q0 + l16) * kD;
  v16bf aq0, aq1;
#pragma unroll
  for (int a = 0; a < 8; ++a) {
    const int dk = 2 * a + 16 * h;
    aq0[2 * a]     = (__bf16)qrow[dk];
    aq0[2 * a + 1] = (__bf16)qrow[dk + 1];
    aq1[2 * a]     = (__bf16)qrow[32 + dk];
    aq1[2 * a + 1] = (__bf16)qrow[32 + dk + 1];
  }

  float mrow[8], lrow[8];
  v8f acc0, acc1, acc2, acc3;
#pragma unroll
  for (int i = 0; i < 8; ++i) {
    mrow[i] = kNegInf; lrow[i] = 0.f;
    acc0[i] = 0.f; acc1[i] = 0.f; acc2[i] = 0.f; acc3[i] = 0.f;
  }

  const uint64_t mask_base = (uint64_t)(uintptr_t)(mask + ((size_t)b * kQ + qwg) * (size_t)kK);
  const unsigned sM_off    = (unsigned)(uintptr_t)&sM[0];   // LDS aperture: low 32 = offset

#pragma unroll 1
  for (int kb = 0; kb < kK; kb += kKTile) {
    // ---- async: TDM pulls the strided 64x32B mask sub-tile into LDS ----
    if (wave == 0) {
      tdm_load_2d_bytes(sM_off, mask_base + (unsigned)kb,
                        /*tile_w=*/kKTile, /*tile_h=*/kQTile,
                        /*row_stride=*/kK, /*tdim0=*/kK, /*tdim1=*/kQ);
    }

    // ---- cooperative staging: K->sK (row-major bf16), V->sVt (transposed) ----
#pragma unroll
    for (int it = 0; it < (kKTile * kD) / (8 * kThreads); ++it) {
      const int chunk = tid + it * kThreads;   // 8 consecutive floats per chunk
      const int row = chunk >> 3;
      const int cb  = (chunk & 7) * 8;
      const float* kp = k + ((size_t)b * kK + kb + row) * kD + cb;
      const float* vp = v + ((size_t)b * kK + kb + row) * kD + cb;
      const float4 ka = *(const float4*)kp;
      const float4 kc = *(const float4*)(kp + 4);
      const float4 va = *(const float4*)vp;
      const float4 vc = *(const float4*)(vp + 4);
      union { __bf16 hv[8]; uint4 u4; } pk;
      pk.hv[0] = (__bf16)ka.x; pk.hv[1] = (__bf16)ka.y;
      pk.hv[2] = (__bf16)ka.z; pk.hv[3] = (__bf16)ka.w;
      pk.hv[4] = (__bf16)kc.x; pk.hv[5] = (__bf16)kc.y;
      pk.hv[6] = (__bf16)kc.z; pk.hv[7] = (__bf16)kc.w;
      *(uint4*)&sK[row * SKS + cb] = pk.u4;            // ds_store_b128
      const __bf16 vb[8] = {(__bf16)va.x, (__bf16)va.y, (__bf16)va.z, (__bf16)va.w,
                            (__bf16)vc.x, (__bf16)vc.y, (__bf16)vc.z, (__bf16)vc.w};
#pragma unroll
      for (int i = 0; i < 8; ++i) sVt[(cb + i) * SVS + row] = vb[i];
      if (kb + kKTile < kK) {
        __builtin_prefetch(kp + kKTile * kD, 0, 1);    // global_prefetch_b8
        __builtin_prefetch(vp + kKTile * kD, 0, 1);
      }
    }
    __builtin_amdgcn_s_wait_tensorcnt(0);   // wave0: TDM done; others: no-op
    __syncthreads();

    // ---- S = Q * K^T : two 16x16 C tiles over 32 keys, 4 bf16 WMMAs ----
    v8f s0, s1;
#pragma unroll
    for (int i = 0; i < 8; ++i) { s0[i] = 0.f; s1[i] = 0.f; }
    {
      v8u bu;
#pragma unroll
      for (int vv = 0; vv < 8; ++vv)
        bu[vv] = *(const unsigned int*)&sK[l16 * SKS + 2 * vv + 16 * h];
      s0 = __builtin_amdgcn_wmma_f32_16x16x32_bf16(false, aq0, false,
              __builtin_bit_cast(v16bf, bu), (short)0, s0, false, false);
#pragma unroll
      for (int vv = 0; vv < 8; ++vv)
        bu[vv] = *(const unsigned int*)&sK[l16 * SKS + 32 + 2 * vv + 16 * h];
      s0 = __builtin_amdgcn_wmma_f32_16x16x32_bf16(false, aq1, false,
              __builtin_bit_cast(v16bf, bu), (short)0, s0, false, false);
#pragma unroll
      for (int vv = 0; vv < 8; ++vv)
        bu[vv] = *(const unsigned int*)&sK[(16 + l16) * SKS + 2 * vv + 16 * h];
      s1 = __builtin_amdgcn_wmma_f32_16x16x32_bf16(false, aq0, false,
              __builtin_bit_cast(v16bf, bu), (short)0, s1, false, false);
#pragma unroll
      for (int vv = 0; vv < 8; ++vv)
        bu[vv] = *(const unsigned int*)&sK[(16 + l16) * SKS + 32 + 2 * vv + 16 * h];
      s1 = __builtin_amdgcn_wmma_f32_16x16x32_bf16(false, aq1, false,
              __builtin_bit_cast(v16bf, bu), (short)0, s1, false, false);
    }

    // ---- mask (from LDS) + online softmax (row = v + 8h, col = lane&15) ----
#pragma unroll
    for (int vv = 0; vv < 8; ++vv) {
      const int mr = (wave * 16 + vv + 8 * h) * kKTile + l16;
      float x0 = sM[mr]      ? kNegInf : s0[vv] * kScale;
      float x1 = sM[mr + 16] ? kNegInf : s1[vv] * kScale;
      float t = fmaxf(x0, x1);
      t = fmaxf(t, __shfl_xor(t, 1, 16));
      t = fmaxf(t, __shfl_xor(t, 2, 16));
      t = fmaxf(t, __shfl_xor(t, 4, 16));
      t = fmaxf(t, __shfl_xor(t, 8, 16));
      const float mnew = fmaxf(mrow[vv], t);
      const float corr = exp2f((mrow[vv] - mnew) * kLog2e);
      const float p0 = exp2f((x0 - mnew) * kLog2e);
      const float p1 = exp2f((x1 - mnew) * kLog2e);
      float rs = p0 + p1;
      rs += __shfl_xor(rs, 1, 16);
      rs += __shfl_xor(rs, 2, 16);
      rs += __shfl_xor(rs, 4, 16);
      rs += __shfl_xor(rs, 8, 16);
      lrow[vv] = lrow[vv] * corr + rs;
      mrow[vv] = mnew;
      acc0[vv] *= corr; acc1[vv] *= corr; acc2[vv] *= corr; acc3[vv] *= corr;
      const int r = vv + 8 * h;
      sP[wave][r * kKTile + l16]      = (__bf16)p0;
      sP[wave][r * kKTile + 16 + l16] = (__bf16)p1;
    }
    // wave-private P transpose through LDS: force store->load ordering
    asm volatile("s_wait_dscnt 0" ::: "memory");

    // ---- O += P * V : 4 bf16 WMMAs over the four 16-wide d tiles ----
    v8u pu;
#pragma unroll
    for (int a = 0; a < 8; ++a) {
      const int kk = 2 * a + 16 * h;
      pu[a] = *(const unsigned int*)&sP[wave][l16 * kKTile + kk];
    }
    const v16bf pa = __builtin_bit_cast(v16bf, pu);
#pragma unroll
    for (int dt = 0; dt < 4; ++dt) {
      v8u bu;
#pragma unroll
      for (int vv = 0; vv < 8; ++vv)
        bu[vv] = *(const unsigned int*)&sVt[(dt * 16 + l16) * SVS + 2 * vv + 16 * h];
      const v16bf bv = __builtin_bit_cast(v16bf, bu);
      v8f* accp = dt == 0 ? &acc0 : dt == 1 ? &acc1 : dt == 2 ? &acc2 : &acc3;
      *accp = __builtin_amdgcn_wmma_f32_16x16x32_bf16(false, pa, false, bv,
                                                      (short)0, *accp, false, false);
    }
    __syncthreads();   // protect sK/sVt/sM before next staging pass
  }

  // ---- epilogue: normalize by softmax denominator and store fp32 ----
#pragma unroll
  for (int dt = 0; dt < 4; ++dt) {
    const v8f a = dt == 0 ? acc0 : dt == 1 ? acc1 : dt == 2 ? acc2 : acc3;
#pragma unroll
    for (int vv = 0; vv < 8; ++vv) {
      out[((size_t)b * kQ + q0 + vv + 8 * h) * kD + dt * 16 + l16] = a[vv] / lrow[vv];
    }
  }
}

extern "C" void kernel_launch(void* const* d_in, const int* in_sizes, int n_in,
                              void* d_out, int out_size, void* d_ws, size_t ws_size,
                              hipStream_t stream) {
  const float*   q    = (const float*)d_in[0];
  const float*   k    = (const float*)d_in[1];
  const float*   v    = (const float*)d_in[2];
  const uint8_t* mask = (const uint8_t*)d_in[3];   // jnp.bool_ -> 1 byte/elem
  float* out = (float*)d_out;
  (void)in_sizes; (void)n_in; (void)out_size; (void)d_ws; (void)ws_size;

  dim3 grid(kB * (kQ / kQTile));   // 16 * 32 = 512 workgroups
  dim3 block(kThreads);            // 128 threads = 4 waves (wave32)
  hipLaunchKernelGGL(fattn_fwd, grid, block, 0, stream, q, k, v, mask, out);
}
